// TwoBodySplineScalarEmbed_47614007444164
// MI455X (gfx1250) — matrix operand order repaired and checked by reading, not compiled, not run on hardware
//
#include <hip/hip_runtime.h>

#define NUM_TYPES    4
#define NUM_SPLINES  16
#define SPLINE_SPAN  12
#define NUM_CHANNELS 64
#define KFULL        256   // 16 classes * 16 splines

typedef __attribute__((ext_vector_type(16))) _Float16 v16h;
typedef __attribute__((ext_vector_type(2)))  _Float16 v2h;
typedef __attribute__((ext_vector_type(8)))  float    v8f;
typedef __attribute__((ext_vector_type(4)))  unsigned uint4v;

union AMat { v16h v; unsigned u[8]; };
union BMat { v16h v; uint4v  q[2]; };
union Pk   { v2h  h; unsigned u;   };

// out[e, c] = sum_s basis(e, s) * W[class_e, s, c]
// Expanded: out = A[E x 256] @ Wflat[256 x 64], A[e, cls*16+s] = basis(e,s), else 0.
// One wave handles a 16-edge tile: 8 K-chunks x 4 N-tiles of v_wmma_f32_16x16x32_f16.
__global__ __launch_bounds__(256)
void TwoBodySplineScalarEmbed_kernel(const int* __restrict__ atom_types,
                                     const int* __restrict__ edge_src,
                                     const int* __restrict__ edge_dst,
                                     const float* __restrict__ norm_length,
                                     const float* __restrict__ class_weight,
                                     float* __restrict__ out,
                                     int n_edges, int n_tiles)
{
    // Wt[c][k] : transposed f16 weights so B tiles are contiguous-in-k per lane.
    __shared__ _Float16 Wt[NUM_CHANNELS * KFULL];   // 32 KB

    const int tid = threadIdx.x;

    // Stage weights: class_weight flat index = k*64 + c  (k = cls*16 + s)
    for (int i = tid; i < NUM_CHANNELS * KFULL; i += blockDim.x) {
        const int k = i >> 6;      // 0..255
        const int c = i & 63;      // 0..63
        Wt[c * KFULL + k] = (_Float16)class_weight[i];
    }
    __syncthreads();

    const int lane = tid & 31;
    const int ln   = lane & 15;   // edge-in-tile (A rows) / channel-in-ntile (B cols)
    const int hi   = lane >> 4;   // lane-group: selects K/s halves
    const int nWaves = (gridDim.x * blockDim.x) >> 5;
    const int gwave  = (blockIdx.x * blockDim.x + tid) >> 5;

    // B tile lane base (halfword units): c = nt*16+ln, k-chunk base kc*32, hi picks K 8..15 / 24..31
    const int ldsLane = ln * KFULL + hi * 8;

    const float grid_scale = (float)(NUM_SPLINES - 1);      // x / h
    const float span_scale = 2.0f / (float)SPLINE_SPAN;

    for (int tile = gwave; tile < n_tiles; tile += nWaves) {
        const int eBase = tile * 16;
        const int e     = eBase + ln;          // both lane halves mirror the same edge

        // Prefetch next tile's edge stream while we compute (gfx1250 global_prefetch).
        const int pe = eBase + nWaves * 16 + ln;
        if (pe < n_edges) {
            __builtin_prefetch(&norm_length[pe], 0, 0);
            __builtin_prefetch(&edge_src[pe], 0, 0);
            __builtin_prefetch(&edge_dst[pe], 0, 0);
        }

        float x = 0.0f;
        int   cls = 0;
        if (e < n_edges) {
            x = norm_length[e];
            const int i0 = edge_src[e];
            const int i1 = edge_dst[e];
            cls = atom_types[i0] * NUM_TYPES + atom_types[i1];
        }

        // 8 basis values per lane: s = hi*8 + j   (matches A-operand K split)
        // basis = cos^2(pi/2 * u), |u|<1 ; v_cos_f32 takes revolutions -> arg u/4
        Pk P[4];
        const float t  = x * grid_scale;
        const int   s0 = hi * 8;
        #pragma unroll
        for (int j = 0; j < 4; ++j) {
            const float u0 = (t - (float)(s0 + 2 * j))     * span_scale;
            const float u1 = (t - (float)(s0 + 2 * j + 1)) * span_scale;
            const float c0 = __builtin_amdgcn_cosf(0.25f * u0);
            const float c1 = __builtin_amdgcn_cosf(0.25f * u1);
            const float b0 = (__builtin_fabsf(u0) < 1.0f) ? c0 * c0 : 0.0f;
            const float b1 = (__builtin_fabsf(u1) < 1.0f) ? c1 * c1 : 0.0f;
            P[j].h = (v2h){(_Float16)b0, (_Float16)b1};
        }

        v8f acc[4];
        #pragma unroll
        for (int nt = 0; nt < 4; ++nt) acc[nt] = (v8f){0.f,0.f,0.f,0.f,0.f,0.f,0.f,0.f};

        #pragma unroll
        for (int kc = 0; kc < 8; ++kc) {
            // A: VGPR 0-3 -> class 2*kc, VGPR 4-7 -> class 2*kc+1 (s from lane group)
            AMat A;
            const unsigned m0 = (cls == 2 * kc)     ? 0xFFFFFFFFu : 0u;
            const unsigned m1 = (cls == 2 * kc + 1) ? 0xFFFFFFFFu : 0u;
            #pragma unroll
            for (int j = 0; j < 4; ++j) {
                A.u[j]     = P[j].u & m0;
                A.u[4 + j] = P[j].u & m1;
            }
            #pragma unroll
            for (int nt = 0; nt < 4; ++nt) {
                const _Float16* bp = &Wt[ldsLane + nt * (16 * KFULL) + kc * 32];
                BMat B;
                B.q[0] = *(const uint4v*)(bp);      // K 0..7  (or 8..15 for hi lanes)
                B.q[1] = *(const uint4v*)(bp + 16); // K 16..23 (or 24..31)
                acc[nt] = __builtin_amdgcn_wmma_f32_16x16x32_f16(
                    /*neg_a=*/false, A.v, /*neg_b=*/false, B.v,
                    /*c_mod=*/(short)0, acc[nt],
                    /*reuse_a=*/false, /*reuse_b=*/false);
            }
        }

        // D layout: VGPR r -> M = r + hi*8 (edge row), N = ln (channel in n-tile)
        #pragma unroll
        for (int nt = 0; nt < 4; ++nt) {
            #pragma unroll
            for (int r = 0; r < 8; ++r) {
                const int ee = eBase + r + hi * 8;
                if (ee < n_edges)
                    out[ee * NUM_CHANNELS + nt * 16 + ln] = acc[nt][r];
            }
        }
    }
}

extern "C" void kernel_launch(void* const* d_in, const int* in_sizes, int n_in,
                              void* d_out, int out_size, void* d_ws, size_t ws_size,
                              hipStream_t stream) {
    const int*   atom_types   = (const int*)d_in[0];
    const int*   edge_index   = (const int*)d_in[1];   // [2, E] flat
    const float* norm_length  = (const float*)d_in[2];
    const float* class_weight = (const float*)d_in[3]; // [16,16,64] flat
    float*       out          = (float*)d_out;

    const int n_edges = in_sizes[2];
    const int n_tiles = (n_edges + 15) / 16;
    const int* edge_src = edge_index;
    const int* edge_dst = edge_index + n_edges;

    const int threads = 256;                 // 8 waves
    const int wavesPerBlock = threads / 32;
    int blocks = (n_tiles + wavesPerBlock - 1) / wavesPerBlock;
    if (blocks > 2048) blocks = 2048;        // grid-stride over tiles

    TwoBodySplineScalarEmbed_kernel<<<blocks, threads, 0, stream>>>(
        atom_types, edge_src, edge_dst, norm_length, class_weight,
        out, n_edges, n_tiles);
}
